// CrossAttention_78872779423998
// MI455X (gfx1250) — matrix operand order repaired and checked by reading, not compiled, not run on hardware
//
#include <hip/hip_runtime.h>

// Cross-attention with 2D-KNN gather (B=4, Q=Z=4096, D=256, H=8, K=16, DH=32)
// Pipeline: posenc(bf16) -> KNN top-16 -> qh GEMM (WMMA bf16) ->
//           fused gather+K/V-projection(WMMA)+attention (2 queries/block) ->
//           out GEMM (WMMA).  Workspace use: ~50 MB.

#define B_  4
#define Q_  4096
#define Z_  4096
#define D_  256
#define H_  8
#define K_  16
#define DH_ 32
#define AP_ 264   // padded LDS row pitch (u16) for gathered rows: 528B = 33*16B

typedef unsigned short u16;
typedef __attribute__((ext_vector_type(16))) __bf16 v16bf;
typedef __attribute__((ext_vector_type(8)))  float  v8f;

union FragBF { v16bf v; uint4 u[2]; };

__device__ __forceinline__ u16 f32_to_bf16(float f) {
  unsigned int u = __float_as_uint(f);
  u += 0x7fffu + ((u >> 16) & 1u);          // round-to-nearest-even
  return (u16)(u >> 16);
}

// ---------------------------------------------------------------------------
// Positional encoding: out_bf16[i] = bf16( x[i] + {sin|cos}(2*pi*(pos @ w_rff)) )
// ---------------------------------------------------------------------------
__global__ void posenc_kernel(const float* __restrict__ x,
                              const float* __restrict__ pos,
                              const float* __restrict__ w_rff,
                              u16* __restrict__ out, int total) {
  int i = blockIdx.x * blockDim.x + threadIdx.x;
  if (i >= total) return;
  int d  = i & (D_ - 1);
  int bn = i >> 8;
  float px = pos[bn * 2 + 0];
  float py = pos[bn * 2 + 1];
  int j = d & 127;
  float pr = 6.283185307179586f * (px * w_rff[j] + py * w_rff[128 + j]);
  float v = (d < 128) ? __sinf(pr) : __cosf(pr);
  out[i] = f32_to_bf16(x[i] + v);
}

// ---------------------------------------------------------------------------
// Weight conversion fp32 -> bf16: [wq|wk|wv|out_w], each 256x256 row-major
// ---------------------------------------------------------------------------
__global__ void cvt_w_kernel(const float* __restrict__ in_proj_w,
                             const float* __restrict__ out_w,
                             u16* __restrict__ wbf) {
  int i = blockIdx.x * blockDim.x + threadIdx.x;
  if (i >= 4 * D_ * D_) return;
  float v = (i < 3 * D_ * D_) ? in_proj_w[i] : out_w[i - 3 * D_ * D_];
  wbf[i] = f32_to_bf16(v);
}

// ---------------------------------------------------------------------------
// KNN: one wave (32 lanes) per query; per-lane sorted top-16 over a strided
// 128-candidate subset, then 16-round wave argmin merge via LDS.
// ---------------------------------------------------------------------------
__global__ __launch_bounds__(128) void knn_kernel(const float* __restrict__ qpos,
                                                  const float* __restrict__ cpos,
                                                  int* __restrict__ knn) {
  __shared__ float sd[4][32 * K_];
  __shared__ int   si[4][32 * K_];
  int wave = threadIdx.x >> 5;
  int lane = threadIdx.x & 31;
  int bq = blockIdx.x * 4 + wave;            // < B_*Q_
  int b  = bq >> 12;
  float qx = qpos[bq * 2 + 0];
  float qy = qpos[bq * 2 + 1];
  const float* cp = cpos + (size_t)b * Z_ * 2;

  float dist[K_]; int idx[K_];
  #pragma unroll
  for (int t = 0; t < K_; ++t) { dist[t] = 3.0e38f; idx[t] = 0; }

  for (int z = lane; z < Z_; z += 32) {
    float2 c = *(const float2*)(cp + z * 2);
    float dx = qx - c.x, dy = qy - c.y;
    float dd = dx * dx + dy * dy;
    if (dd < dist[K_ - 1]) {                 // threshold-guarded sorted insert
      float cd = dd; int ci = z;
      #pragma unroll
      for (int t = 0; t < K_; ++t) {
        bool sm = cd < dist[t];
        float td = dist[t]; int ti = idx[t];
        dist[t] = sm ? cd : dist[t];
        idx[t]  = sm ? ci : idx[t];
        cd = sm ? td : cd;
        ci = sm ? ti : ci;
      }
    }
  }
  #pragma unroll
  for (int t = 0; t < K_; ++t) {
    sd[wave][lane * K_ + t] = dist[t];
    si[wave][lane * K_ + t] = idx[t];
  }
  int p = 0;
  int* outp = knn + bq * K_;
  for (int r = 0; r < K_; ++r) {
    float h = (p < K_) ? sd[wave][lane * K_ + p] : 3.0e38f;
    float bd = h; int bl = lane;
    #pragma unroll
    for (int off = 16; off >= 1; off >>= 1) {
      float od = __shfl_xor(bd, off, 32);
      int   ol = __shfl_xor(bl, off, 32);
      if (od < bd) { bd = od; bl = ol; }
    }
    if (lane == bl) {
      outp[r] = si[wave][lane * K_ + p];
      ++p;
    }
  }
}

// ---------------------------------------------------------------------------
// GEMM: C[M x 256] = bf16A[M x 256] @ bf16W[n][k]^T + bias, f32 out.
// One wave computes a 16x64 strip (4 accumulators), K-loop of 8x WMMA 16x16x32.
// ---------------------------------------------------------------------------
__global__ __launch_bounds__(256) void gemm_bias_kernel(const u16* __restrict__ A,
                                                        const u16* __restrict__ W,
                                                        const float* __restrict__ bias,
                                                        float* __restrict__ C, int M) {
  int gwave = (int)((blockIdx.x * 256u + threadIdx.x) >> 5);
  int lane  = threadIdx.x & 31;
  int nstrip = gwave & 3;                    // 4 strips of 64 cols
  int mtile  = gwave >> 2;
  if (mtile * 16 >= M) return;
  int hi = lane >> 4;
  int am = lane & 15;
  const u16* arow = A + (size_t)(mtile * 16 + am) * D_;

  v8f acc[4] = {};
  for (int ks = 0; ks < 8; ++ks) {
    int k0 = ks * 32;
    FragBF a;
    a.u[0] = *(const uint4*)(arow + k0 + 8 * hi);
    a.u[1] = *(const uint4*)(arow + k0 + 16 + 8 * hi);
    #pragma unroll
    for (int tt = 0; tt < 4; ++tt) {
      int n0 = nstrip * 64 + tt * 16;
      const u16* wrow = W + (size_t)(n0 + am) * D_ + k0;
      FragBF bf;
      bf.u[0] = *(const uint4*)(wrow + 8 * hi);
      bf.u[1] = *(const uint4*)(wrow + 16 + 8 * hi);
      acc[tt] = __builtin_amdgcn_wmma_f32_16x16x32_bf16(
          false, a.v, false, bf.v, (short)0, acc[tt], false, false);
    }
  }
  #pragma unroll
  for (int tt = 0; tt < 4; ++tt) {
    int n = nstrip * 64 + tt * 16 + am;
    float bv = bias[n];
    #pragma unroll
    for (int j = 0; j < 8; ++j)
      C[(size_t)(mtile * 16 + 8 * hi + j) * D_ + n] = acc[tt][j] + bv;
  }
}

// ---------------------------------------------------------------------------
// Fused per-query kernel, 2 queries per block:
//  - stage the 2x16 gathered neighbor rows into LDS once (coalesced),
//  - each wave computes 4 N-tiles for BOTH queries: every weight B-fragment
//    loaded from L2 feeds 2 WMMAs (halves weight re-fetch traffic),
//  - A-fragments come from LDS (padded pitch -> conflict-free ds_load_b128),
//  - then 8-head softmax attention per query.
// Block = 256 threads (8 waves); waves 0-3 -> kh tiles, waves 4-7 -> vh tiles.
// ---------------------------------------------------------------------------
__global__ __launch_bounds__(256) void kv_attn_kernel(const u16* __restrict__ cfeat,
                                                      const u16* __restrict__ wk,
                                                      const u16* __restrict__ wv,
                                                      const float* __restrict__ bk,
                                                      const float* __restrict__ bv,
                                                      const float* __restrict__ qh,
                                                      const int* __restrict__ knn,
                                                      u16* __restrict__ obf) {
  __shared__ u16   a_lds[2][K_][AP_];        // gathered bf16 rows, padded pitch
  __shared__ float kh_lds[2][K_][D_];
  __shared__ float vh_lds[2][K_][D_];
  __shared__ float qrow[2][D_];
  __shared__ float attn_lds[2][H_][K_];
  __shared__ int   idx_lds[2][K_];

  int bq0  = blockIdx.x * 2;                 // pair never straddles a batch
  int b    = bq0 >> 12;
  int tid  = threadIdx.x;
  int wave = tid >> 5;
  int lane = tid & 31;
  int hi   = lane >> 4;
  int am   = lane & 15;

  if (tid < 2 * K_) {
    int g = tid >> 4;
    idx_lds[g][tid & 15] = knn[(bq0 + g) * K_ + (tid & 15)];
  }
  qrow[0][tid] = qh[(size_t)(bq0 + 0) * D_ + tid];
  qrow[1][tid] = qh[(size_t)(bq0 + 1) * D_ + tid];
  __syncthreads();

  // Stage gathered rows: thread t copies 64B of row r of query g.
  {
    int g = tid >> 7;                        // 0..1
    int r = (tid >> 3) & 15;                 // 0..15
    int c = tid & 7;                         // 0..7, 64B chunks
    const u16* src = cfeat + (size_t)(b * Z_ + idx_lds[g][r]) * D_ + c * 32;
    #pragma unroll
    for (int j = 0; j < 4; ++j)
      *(uint4*)(&a_lds[g][r][c * 32 + j * 8]) = *(const uint4*)(src + j * 8);
  }
  __syncthreads();

  // Projection: each wave owns 4 N-tiles (kh for waves 0-3, vh for 4-7),
  // computed for both queries so each B-fragment is reused twice.
  v8f acc[2][4] = {};
  for (int ks = 0; ks < 8; ++ks) {
    int k0 = ks * 32;
    FragBF a0, a1;
    a0.u[0] = *(const uint4*)(&a_lds[0][am][k0 + 8 * hi]);
    a0.u[1] = *(const uint4*)(&a_lds[0][am][k0 + 16 + 8 * hi]);
    a1.u[0] = *(const uint4*)(&a_lds[1][am][k0 + 8 * hi]);
    a1.u[1] = *(const uint4*)(&a_lds[1][am][k0 + 16 + 8 * hi]);
    #pragma unroll
    for (int tt = 0; tt < 4; ++tt) {
      int t = wave * 4 + tt;                 // 0..15 -> kh, 16..31 -> vh
      const u16* W = (t < 16) ? wk : wv;
      int n0 = (t & 15) * 16;
      const u16* wrow = W + (size_t)(n0 + am) * D_ + k0;
      FragBF bf;
      bf.u[0] = *(const uint4*)(wrow + 8 * hi);
      bf.u[1] = *(const uint4*)(wrow + 16 + 8 * hi);
      acc[0][tt] = __builtin_amdgcn_wmma_f32_16x16x32_bf16(
          false, a0.v, false, bf.v, (short)0, acc[0][tt], false, false);
      acc[1][tt] = __builtin_amdgcn_wmma_f32_16x16x32_bf16(
          false, a1.v, false, bf.v, (short)0, acc[1][tt], false, false);
    }
  }
  #pragma unroll
  for (int tt = 0; tt < 4; ++tt) {
    int t = wave * 4 + tt;
    int n = (t & 15) * 16 + am;
    float bvv = (t < 16) ? bk[n] : bv[n];
    #pragma unroll
    for (int g = 0; g < 2; ++g) {
      float (*dst)[D_] = (t < 16) ? kh_lds[g] : vh_lds[g];
      #pragma unroll
      for (int j = 0; j < 8; ++j) dst[8 * hi + j][n] = acc[g][tt][j] + bvv;
    }
  }
  __syncthreads();

  // Attention: wave = head; lanes duplicated across halves for the 16 scores.
  int h = wave;
  int k = lane & 15;
  #pragma unroll
  for (int g = 0; g < 2; ++g) {
    float s = 0.f;
    #pragma unroll
    for (int d = 0; d < DH_; ++d)
      s += qrow[g][h * DH_ + d] * kh_lds[g][k][h * DH_ + d];
    s *= 0.17677669529663687f;               // 1/sqrt(32)
    float mx = s;
    #pragma unroll
    for (int off = 8; off >= 1; off >>= 1) mx = fmaxf(mx, __shfl_xor(mx, off, 32));
    float e = __expf(s - mx);
    float sum = e;
    #pragma unroll
    for (int off = 8; off >= 1; off >>= 1) sum += __shfl_xor(sum, off, 32);
    float aw = e / sum;
    if (lane < 16) attn_lds[g][h][lane] = aw;
  }
  __syncthreads();

  #pragma unroll
  for (int g = 0; g < 2; ++g) {
    float o = 0.f;
    #pragma unroll
    for (int kk = 0; kk < K_; ++kk)
      o += attn_lds[g][h][kk] * vh_lds[g][kk][h * DH_ + lane];
    obf[(size_t)(bq0 + g) * D_ + h * DH_ + lane] = f32_to_bf16(o);
  }
}

// ---------------------------------------------------------------------------
__global__ void copy_pos_kernel(const float* __restrict__ qpos,
                                float* __restrict__ dst, int n) {
  int i = blockIdx.x * blockDim.x + threadIdx.x;
  if (i < n) dst[i] = qpos[i];
}

// ---------------------------------------------------------------------------
extern "C" void kernel_launch(void* const* d_in, const int* in_sizes, int n_in,
                              void* d_out, int out_size, void* d_ws, size_t ws_size,
                              hipStream_t stream) {
  (void)in_sizes; (void)n_in; (void)out_size; (void)ws_size;
  const float* query       = (const float*)d_in[0];
  const float* query_pos   = (const float*)d_in[1];
  const float* context     = (const float*)d_in[2];
  const float* context_pos = (const float*)d_in[3];
  const float* w_rff       = (const float*)d_in[4];
  const float* in_proj_w   = (const float*)d_in[5];
  const float* in_proj_b   = (const float*)d_in[6];
  const float* out_w       = (const float*)d_in[7];
  const float* out_b       = (const float*)d_in[8];
  float* out = (float*)d_out;

  const int BQ = B_ * Q_;                    // 16384
  const int NF = BQ * D_;                    // 4,194,304

  // workspace layout (~50 MB)
  u16*   qfeat = (u16*)d_ws;                 //  8 MB
  u16*   cfeat = qfeat + NF;                 //  8 MB
  u16*   wbf   = cfeat + NF;                 //  0.5 MB (wq|wk|wv|out_w)
  int*   knn   = (int*)(wbf + 4 * D_ * D_);  //  1 MB
  float* qh    = (float*)(knn + BQ * K_);    // 16 MB
  u16*   obf   = (u16*)(qh + NF);            //  8 MB

  posenc_kernel<<<(NF + 255) / 256, 256, 0, stream>>>(query, query_pos, w_rff, qfeat, NF);
  posenc_kernel<<<(NF + 255) / 256, 256, 0, stream>>>(context, context_pos, w_rff, cfeat, NF);
  cvt_w_kernel<<<(4 * D_ * D_ + 255) / 256, 256, 0, stream>>>(in_proj_w, out_w, wbf);
  knn_kernel<<<BQ / 4, 128, 0, stream>>>(query_pos, context_pos, knn);
  gemm_bias_kernel<<<BQ / 32, 256, 0, stream>>>(qfeat, wbf, in_proj_b, qh, BQ);
  kv_attn_kernel<<<BQ / 2, 256, 0, stream>>>(cfeat, wbf + D_ * D_, wbf + 2 * D_ * D_,
                                             in_proj_b + D_, in_proj_b + 2 * D_,
                                             qh, knn, obf);
  gemm_bias_kernel<<<BQ / 32, 256, 0, stream>>>(obf, wbf + 3 * D_ * D_, out_b, out, BQ);
  copy_pos_kernel<<<(BQ * 2 + 255) / 256, 256, 0, stream>>>(query_pos, out + (size_t)NF, BQ * 2);
}
